// AttentionFact_2937757630770
// MI455X (gfx1250) — matrix operand order repaired and checked by reading, not compiled, not run on hardware
//
#include <hip/hip_runtime.h>
#include <math.h>

#define BB 16
#define SS 4096
#define HH 1024
#define CC 64

typedef __attribute__((ext_vector_type(16))) _Float16 v16h;
typedef __attribute__((ext_vector_type(8)))  _Float16 v8h;
typedef __attribute__((ext_vector_type(8)))  float    v8f;
typedef __attribute__((ext_vector_type(4)))  unsigned u32x4;
typedef __attribute__((ext_vector_type(8)))  int      i32x8;
typedef __attribute__((ext_vector_type(4)))  int      i32x4;

#if __has_builtin(__builtin_amdgcn_tensor_load_to_lds)
#define HAVE_TDM 1
#else
#define HAVE_TDM 0
#endif

// Device-pass-only diagnostics so the success stderr tells us which data-mover
// path actually lowered (host pass would always lack amdgcn builtins).
#if defined(__AMDGCN__)
#if HAVE_TDM
#if __clang_major__ >= 23
#pragma message("CDNA5: tensor_load_to_lds builtin available (6-arg form)")
#else
#pragma message("CDNA5: tensor_load_to_lds builtin available (5-arg form)")
#endif
#else
#warning "CDNA5: tensor_load_to_lds builtin NOT available; vector-copy fallback in use"
#endif
#if __has_builtin(__builtin_amdgcn_s_wait_tensorcnt)
#pragma message("CDNA5: s_wait_tensorcnt builtin available")
#else
#warning "CDNA5: s_wait_tensorcnt builtin NOT available; inline asm fallback in use"
#endif
#endif

static __device__ __forceinline__ void wait_tensorcnt0() {
#if __has_builtin(__builtin_amdgcn_s_wait_tensorcnt)
  __builtin_amdgcn_s_wait_tensorcnt(0);
#else
  asm volatile("s_wait_tensorcnt 0x0" ::: "memory");
#endif
}

static __device__ __forceinline__ v8h cvt8h(float4 a, float4 b) {
  v8h r;
  r[0] = (_Float16)a.x; r[1] = (_Float16)a.y; r[2] = (_Float16)a.z; r[3] = (_Float16)a.w;
  r[4] = (_Float16)b.x; r[5] = (_Float16)b.y; r[6] = (_Float16)b.z; r[7] = (_Float16)b.w;
  return r;
}

static __device__ __forceinline__ v16h cat16h(v8h lo, v8h hi) {
  return __builtin_shufflevector(lo, hi, 0,1,2,3,4,5,6,7,8,9,10,11,12,13,14,15);
}

static __device__ __forceinline__ v8f wmma_f16(v16h a, v16h b, v8f c) {
  return __builtin_amdgcn_wmma_f32_16x16x32_f16(false, a, false, b, (short)0, c,
                                                false, false);
}

#if HAVE_TDM
// One TDM descriptor: copy a 64(row) x 128(col) f16 tile out of a [64][1024]
// f16 tensor (col offset baked into global_addr) into LDS with 4-DWORD padding
// after every 64 data DWORDs -> LDS row stride 136 f16 (bank-conflict free).
static __device__ __forceinline__ void tdm_load_q_chunk(unsigned lds_addr,
                                                        const _Float16* gsrc) {
  unsigned long long ga = (unsigned long long)(size_t)gsrc;
  u32x4 g0;
  g0[0] = 1u;                                             // count=1 user D#
  g0[1] = lds_addr;                                       // LDS byte address
  g0[2] = (unsigned)ga;                                   // global_addr[31:0]
  g0[3] = (unsigned)((ga >> 32) & 0x01ffffffull) | (2u << 30); // addr[56:32], type=2
  i32x8 g1;
  g1[0] = (1 << 16)      // data_size = 1 -> 2 bytes
        | (1 << 20)      // pad_enable
        | (5 << 22)      // pad_interval: 64 DWORDs (one 128-f16 row)
        | (3 << 25);     // pad_amount: 4 DWORDs (8 f16) -> stride 136
  g1[1] = (int)((HH & 0xffff) << 16);                     // tensor_dim0 lo16
  g1[2] = (int)(((unsigned)HH >> 16) | ((CC & 0xffff) << 16)); // td0 hi, td1 lo
  g1[3] = (int)(((unsigned)CC >> 16) | (128u << 16));     // td1 hi, tile_dim0=128
  g1[4] = CC;                                             // tile_dim1=64, tile_dim2=0
  g1[5] = HH;                                             // tensor_dim0_stride lo32
  g1[6] = 0;
  g1[7] = 0;
  i32x4 z4 = {0, 0, 0, 0};
#if __clang_major__ >= 23
  i32x8 z8 = {0, 0, 0, 0, 0, 0, 0, 0};
  __builtin_amdgcn_tensor_load_to_lds(g0, g1, z4, z4, z8, 0);
#else
  __builtin_amdgcn_tensor_load_to_lds(g0, g1, z4, z4, 0);
#endif
}
#endif

// -------- Phase 0: querys fp32 -> f16 (done once, 128KB) -----------------------
__global__ __launch_bounds__(256) void k_prep(const float* __restrict__ querys,
                                              _Float16* __restrict__ qh) {
  const int i = (blockIdx.x * 256 + threadIdx.x) * 8;
  const float4* src = (const float4*)(querys + i);
  float4 a = src[0], b = src[1];
  *(v8h*)(qh + i) = cvt8h(a, b);
}

// -------- Phase 1: scores[b][c][s] = hidden[b,s,:] . querys[c,:] ---------------
__global__ __launch_bounds__(256) void k_scores(const float* __restrict__ hidden,
                                                const _Float16* __restrict__ qh,
                                                float* __restrict__ scores) {
  // union: f16 q-tile [64][136] (17.4KB) during K loop, f32 staging [64][132]
  // (33.8KB) for the coalesced epilogue write.
  __shared__ __align__(16) unsigned char smem[CC * 132 * sizeof(float)];
  _Float16 (*q)[136] = reinterpret_cast<_Float16 (*)[136]>(smem);
  float    (*st)[132] = reinterpret_cast<float (*)[132]>(smem);

  const int b    = blockIdx.y;
  const int s0   = blockIdx.x * 256;
  const int tid  = threadIdx.x;
  const int wav  = tid >> 5;
  const int lane = tid & 31;
  const int l16  = lane & 15;
  const int hi   = lane >> 4;

  // two A fragments per wave: rows s0+32*wav+l16 and +16
  const float* hrow0 = hidden + ((size_t)b * SS + (size_t)(s0 + wav * 32 + l16)) * HH;
  const float* hrow1 = hrow0 + (size_t)16 * HH;

  v8f acc[4][2] = {};

  for (int kk = 0; kk < HH; kk += 128) {
    __syncthreads();
#if HAVE_TDM
    if (wav == 0) {
      tdm_load_q_chunk((unsigned)(size_t)&q[0][0], qh + kk);
      wait_tensorcnt0();
    }
#else
    {
      const int qc = tid >> 2, qo = (tid & 3) * 32;
      const v8h* src = (const v8h*)(qh + (size_t)qc * HH + kk + qo);
      v8h* dst = (v8h*)&q[qc][qo];
#pragma unroll
      for (int u = 0; u < 4; ++u) dst[u] = src[u];
    }
#endif
    // warm the next hidden K-chunk while this one computes
    if (kk + 128 < HH) {
      __builtin_prefetch(hrow0 + kk + 128, 0, 1);
      __builtin_prefetch(hrow1 + kk + 128, 0, 1);
    }
    __syncthreads();
#pragma unroll
    for (int k2 = 0; k2 < 4; ++k2) {
      const int k = kk + k2 * 32;
      // issue all 8 A loads into distinct temps, then convert
      const float4* a0p = (const float4*)(hrow0 + k + hi * 8);
      const float4* a0q = (const float4*)(hrow0 + k + 16 + hi * 8);
      const float4* a1p = (const float4*)(hrow1 + k + hi * 8);
      const float4* a1q = (const float4*)(hrow1 + k + 16 + hi * 8);
      float4 t0 = a0p[0], t1 = a0p[1], t2 = a0q[0], t3 = a0q[1];
      float4 u0 = a1p[0], u1 = a1p[1], u2 = a1q[0], u3 = a1q[1];
      v16h aA = cat16h(cvt8h(t0, t1), cvt8h(t2, t3));
      v16h aB = cat16h(cvt8h(u0, u1), cvt8h(u2, u3));
      const int boff = k2 * 32 + hi * 8;
#pragma unroll
      for (int j = 0; j < 4; ++j) {
        v16h bj = cat16h(*(const v8h*)&q[j * 16 + l16][boff],
                         *(const v8h*)&q[j * 16 + l16][boff + 16]);
        acc[j][0] = wmma_f16(aA, bj, acc[j][0]);
        acc[j][1] = wmma_f16(aB, bj, acc[j][1]);
      }
    }
  }
  // epilogue: two s-half passes through LDS for fully coalesced global stores
#pragma unroll
  for (int p = 0; p < 2; ++p) {
    __syncthreads();
    if ((wav >> 2) == p) {
      const int wl = wav & 3;
#pragma unroll
      for (int f = 0; f < 2; ++f)
#pragma unroll
        for (int r = 0; r < 8; ++r) {
          const int srow = wl * 32 + f * 16 + hi * 8 + r;
#pragma unroll
          for (int j = 0; j < 4; ++j)
            st[j * 16 + l16][srow] = acc[j][f][r];
        }
    }
    __syncthreads();
    const int c = tid >> 2, off = (tid & 3) * 32;
    float* orow = scores + ((size_t)(b * CC + c)) * SS + s0 + p * 128 + off;
#pragma unroll
    for (int i = 0; i < 8; ++i)
      *(float4*)(orow + 4 * i) = *(const float4*)&st[c][off + 4 * i];
  }
}

// -------- Phase 2: softmax stats + f16 probabilities ---------------------------
__global__ __launch_bounds__(256) void k_stats(const float* __restrict__ scores,
                                               float* __restrict__ lrow,
                                               _Float16* __restrict__ ph) {
  __shared__ float red[256];
  const int bc  = blockIdx.x;
  const int tid = threadIdx.x;
  const float* row = scores + (size_t)bc * SS;
  float v[16];
  float mx = -3.4e38f;
#pragma unroll
  for (int i = 0; i < 16; ++i) { v[i] = row[tid + 256 * i]; mx = fmaxf(mx, v[i]); }
  red[tid] = mx;
  __syncthreads();
  for (int s = 128; s > 0; s >>= 1) {
    if (tid < s) red[tid] = fmaxf(red[tid], red[tid + s]);
    __syncthreads();
  }
  const float m = red[0];
  __syncthreads();
  float sum = 0.f;
  _Float16 pv[16];
#pragma unroll
  for (int i = 0; i < 16; ++i) {
    float e = __expf(v[i] - m);
    sum += e;
    pv[i] = (_Float16)e;
  }
  red[tid] = sum;
  __syncthreads();
  for (int s = 128; s > 0; s >>= 1) {
    if (tid < s) red[tid] += red[tid + s];
    __syncthreads();
  }
  _Float16* prow = ph + (size_t)bc * SS;
#pragma unroll
  for (int i = 0; i < 16; ++i) prow[tid + 256 * i] = pv[i];
  if (tid == 0) lrow[bc] = red[0];
}

// -------- Phase 3: out[b][c][h] = (1/l) * sum_s p[c,s] * hidden[s,h] -----------
__global__ __launch_bounds__(256) void k_pool(const float* __restrict__ hidden,
                                              const _Float16* __restrict__ ph,
                                              const float* __restrict__ lrow,
                                              float* __restrict__ out) {
  // two transposed [h][s] f16 slices (K=32 each) per barrier pair, padded rows
  __shared__ __align__(16) _Float16 T[2][64][40];  // 10.2 KB
  __shared__ float linv[CC];

  const int b    = blockIdx.y;
  const int h0   = blockIdx.x * 64;
  const int tid  = threadIdx.x;
  const int wav  = tid >> 5;
  const int lane = tid & 31;
  const int l16  = lane & 15;
  const int hi   = lane >> 4;

  if (tid < CC) linv[tid] = 1.0f / lrow[b * CC + tid];

  const int ci   = wav & 3;
  const int nj0  = (wav >> 2) * 2;
  const int crow = ci * 16 + l16;
  const _Float16* prow = ph + ((size_t)(b * CC + crow)) * SS;

  const int sr = tid >> 3;
  const int sc = (tid & 7) * 8;
  const float* hbase = hidden + (size_t)b * SS * HH + h0;

  v8f acc0 = {}, acc1 = {};

  for (int k = 0; k < SS; k += 64) {
    // issue all global loads (both staging slices + both A fragments) up front
    const float4* hp0 = (const float4*)(hbase + (size_t)(k + sr) * HH + sc);
    const float4* hp1 = (const float4*)(hbase + (size_t)(k + 32 + sr) * HH + sc);
    float4 f0 = hp0[0], f1 = hp0[1];
    float4 g0 = hp1[0], g1 = hp1[1];
    v8h pa0 = *(const v8h*)(prow + k + hi * 8);
    v8h pb0 = *(const v8h*)(prow + k + 16 + hi * 8);
    v8h pa1 = *(const v8h*)(prow + k + 32 + hi * 8);
    v8h pb1 = *(const v8h*)(prow + k + 48 + hi * 8);

    T[0][sc + 0][sr] = (_Float16)f0.x;
    T[0][sc + 1][sr] = (_Float16)f0.y;
    T[0][sc + 2][sr] = (_Float16)f0.z;
    T[0][sc + 3][sr] = (_Float16)f0.w;
    T[0][sc + 4][sr] = (_Float16)f1.x;
    T[0][sc + 5][sr] = (_Float16)f1.y;
    T[0][sc + 6][sr] = (_Float16)f1.z;
    T[0][sc + 7][sr] = (_Float16)f1.w;
    T[1][sc + 0][sr] = (_Float16)g0.x;
    T[1][sc + 1][sr] = (_Float16)g0.y;
    T[1][sc + 2][sr] = (_Float16)g0.z;
    T[1][sc + 3][sr] = (_Float16)g0.w;
    T[1][sc + 4][sr] = (_Float16)g1.x;
    T[1][sc + 5][sr] = (_Float16)g1.y;
    T[1][sc + 6][sr] = (_Float16)g1.z;
    T[1][sc + 7][sr] = (_Float16)g1.w;
    __syncthreads();

    const int boff = hi * 8;
#pragma unroll
    for (int u = 0; u < 2; ++u) {
      v16h a = cat16h(u ? pa1 : pa0, u ? pb1 : pb0);
      v16h b0 = cat16h(*(const v8h*)&T[u][nj0 * 16 + l16][boff],
                       *(const v8h*)&T[u][nj0 * 16 + l16][boff + 16]);
      v16h b1 = cat16h(*(const v8h*)&T[u][(nj0 + 1) * 16 + l16][boff],
                       *(const v8h*)&T[u][(nj0 + 1) * 16 + l16][boff + 16]);
      acc0 = wmma_f16(a, b0, acc0);
      acc1 = wmma_f16(a, b1, acc1);
    }
    __syncthreads();
  }

#pragma unroll
  for (int r = 0; r < 8; ++r) {
    const int c = ci * 16 + hi * 8 + r;
    const float s = linv[c];
    out[(size_t)b * (CC * HH) + (size_t)c * HH + h0 + nj0 * 16 + l16] = acc0[r] * s;
    out[(size_t)b * (CC * HH) + (size_t)c * HH + h0 + (nj0 + 1) * 16 + l16] = acc1[r] * s;
  }
}

extern "C" void kernel_launch(void* const* d_in, const int* in_sizes, int n_in,
                              void* d_out, int out_size, void* d_ws, size_t ws_size,
                              hipStream_t stream) {
  (void)in_sizes; (void)n_in; (void)out_size; (void)ws_size;
  const float* hidden = (const float*)d_in[0];
  const float* querys = (const float*)d_in[1];
  float* out = (float*)d_out;

  char* ws = (char*)d_ws;
  float*     scores = (float*)ws;                              // 16 MB
  float*     lrow   = (float*)(ws + 16777216);                 // 4 KB
  _Float16*  qh     = (_Float16*)(ws + 16777216 + 4096);       // 128 KB
  _Float16*  ph     = (_Float16*)(ws + 16777216 + 4096 + 131072); // 8 MB

  k_prep<<<dim3(CC * HH / 2048), 256, 0, stream>>>(querys, qh);
  k_scores<<<dim3(SS / 256, BB), 256, 0, stream>>>(hidden, qh, scores);
  k_stats<<<dim3(BB * CC), 256, 0, stream>>>(scores, lrow, ph);
  k_pool<<<dim3(HH / 64, BB), 256, 0, stream>>>(hidden, ph, lrow, out);
}
